// Gemma3Main_3856880632303
// MI455X (gfx1250) — matrix operand order repaired and checked by reading, not compiled, not run on hardware
//
#include <hip/hip_runtime.h>

// ---------------- model constants ----------------
constexpr int L_ = 26, T_ = 512, D_ = 768;
constexpr int NH_ = 12, NG_ = 4, GS_ = 3, HD_ = 64, FF_ = 2048;
constexpr int CACHE_ = 1536, S_ = 2048, V_ = 32000;
constexpr int QKV_ = (NH_ + 2 * NG_) * HD_; // 1280

// ---------------- types ----------------
typedef __bf16 bf16;
typedef __attribute__((ext_vector_type(16))) __bf16 v16bf;
typedef __attribute__((ext_vector_type(8))) float v8f;
typedef __attribute__((ext_vector_type(4))) unsigned int v4u;
typedef __attribute__((ext_vector_type(8))) int v8i;
typedef __attribute__((ext_vector_type(4))) int v4i;

struct U32x8 { uint4 lo; uint4 hi; };

static __device__ __forceinline__ bf16 f2bf(float f) {
  unsigned u = __builtin_bit_cast(unsigned, f);
  unsigned r = u + 0x7FFFu + ((u >> 16) & 1u); // round-to-nearest-even
  unsigned short h = (unsigned short)(r >> 16);
  return __builtin_bit_cast(bf16, h);
}
static __device__ __forceinline__ float bf2f(bf16 b) {
  unsigned short h = __builtin_bit_cast(unsigned short, b);
  unsigned u = ((unsigned)h) << 16;
  return __builtin_bit_cast(float, u);
}
static __device__ __forceinline__ float ldval(const float* p) { return *p; }
static __device__ __forceinline__ float ldval(const bf16* p) { return bf2f(*p); }

// ---------------- TDM: DMA one bf16 tile [rows x 32] (row stride lda elems)
// from global memory into LDS (packed [rows][32], 2B elems).  OOB rows are
// hardware zero-filled.  D# layout per CDNA5 ISA.
// clang-23 toolchain: 6-arg builtin (g0 v4u, g1 v8i, g2 v4i, g3 v4i, g4 v8i, cpol).
static __device__ __forceinline__ void tdm_load_a_tile(const bf16* gsrc,
                                                       unsigned lds_off,
                                                       int rows, int lda) {
  unsigned long long ga = (unsigned long long)(size_t)gsrc;
  v4u g0;
  g0[0] = 1u;                                   // count=1, user-mode, no gather
  g0[1] = lds_off;                              // lds_addr (bytes)
  g0[2] = (unsigned)(ga & 0xFFFFFFFFu);         // global_addr[31:0]
  g0[3] = (unsigned)((ga >> 32) & 0x01FFFFFFu)  // global_addr[56:32]
          | (2u << 30);                         // type=2 ("image")
  v8i g1;
  g1[0] = (int)(1u << 16);                      // data_size=1 (2 bytes)
  g1[1] = (int)(32u << 16);                     // tensor_dim0[15:0]=32 (bits 63:48)
  g1[2] = (int)(((unsigned)rows & 0xFFFFu) << 16); // tensor_dim1[15:0] (bits 95:80)
  g1[3] = (int)((((unsigned)rows >> 16) & 0xFFFFu)) // tensor_dim1[31:16]
          | (int)(32u << 16);                   // tile_dim0=32 (bits 127:112)
  g1[4] = (int)128u;                            // tile_dim1=128; tile_dim2=0
  g1[5] = (int)(unsigned)lda;                   // tensor_dim0_stride[31:0]
  g1[6] = 0;                                    // stride hi, dim1_stride lo
  g1[7] = 0;
  v4i g2 = {0, 0, 0, 0};
  v4i g3 = {0, 0, 0, 0};
  v8i g4 = {0, 0, 0, 0, 0, 0, 0, 0};
  __builtin_amdgcn_tensor_load_to_lds(g0, g1, g2, g3, g4, 0);
}

// ---------------- generic batched WMMA GEMM ----------------
// C[M,N] f32 = A[M,K] bf16  @  B[K,N] (f32 or bf16, converted to bf16 in LDS)
// Block tile 128x128, BK=32, 256 threads = 8 waves, wave tile 64x32 (4x2 wmma).
// Double-buffered LDS; A tiles via Tensor Data Mover, B tiles via VALU
// (fused f32->bf16 convert + [k][n]->[n][k] transpose).  K % 32 == 0.
template <typename TB>
__global__ void __launch_bounds__(256)
gemm_wmma_kernel(const bf16* __restrict__ A, const TB* __restrict__ B,
                 float* __restrict__ C, int M, int N, int K,
                 int lda, int ldb, int ldc,
                 long long sA, long long sB, long long sC) {
  __shared__ bf16 lds_a[2][128 * 32];   // [m][k]
  __shared__ bf16 lds_b[2][128 * 32];   // [n][k]

  const int tid  = threadIdx.x;
  const int lane = tid & 31;
  const int wave = tid >> 5;
  const int wr = wave >> 2;             // 0..1  (rows of 64)
  const int wc = wave & 3;              // 0..3  (cols of 32)
  const int bm = blockIdx.y * 128;
  const int bn = blockIdx.x * 128;
  const int z  = blockIdx.z;

  A += (long long)z * sA;
  B += (long long)z * sB;
  C += (long long)z * sC;

  v8f acc[4][2];
#pragma unroll
  for (int i = 0; i < 4; ++i)
#pragma unroll
    for (int j = 0; j < 2; ++j)
#pragma unroll
      for (int e = 0; e < 8; ++e) acc[i][j][e] = 0.0f;

  // B staging index map (per thread: 16 consecutive columns of one K-row)
  const int bk  = tid >> 3;             // 0..31 (K row)
  const int bn0 = (tid & 7) * 16;       // 0..112 (N base)
  const int arows = (M - bm < 128) ? (M - bm) : 128;

  // fragment gather maps (ISA §7.12.2 VGPR layouts)
  const int fr  = lane & 15;
  const int ak  = (lane >> 4) * 8;      // A: K 0..7 / 8..15 halves
  const int bkk = (lane >> 4) * 8;      // B: K 0..7 / 8..15 halves

  // ---- prologue: stage tile k0=0 into buffer 0 ----
  {
    if (wave == 0)
      tdm_load_a_tile(A + (long long)bm * lda, (unsigned)(size_t)&lds_a[0][0],
                      arows, lda);
    const TB* src = B + (long long)bk * ldb + bn;
    float bv[16];
#pragma unroll
    for (int j = 0; j < 16; ++j) {
      const int gc = bn0 + j;
      bv[j] = (bn + gc < N) ? ldval(src + gc) : 0.0f;
    }
#pragma unroll
    for (int j = 0; j < 16; ++j) lds_b[0][(bn0 + j) * 32 + bk] = f2bf(bv[j]);
    if (wave == 0) __builtin_amdgcn_s_wait_tensorcnt((short)0);
    __syncthreads();
  }

  int cur = 0;
  for (int k0 = 0; k0 < K; k0 += 32) {
    const int nxt = cur ^ 1;
    const bool do_stage = (k0 + 32) < K;

    // ---- kick off next tile: TDM for A, global loads (regs) for B ----
    float bv[16];
    if (do_stage) {
      if (wave == 0)
        tdm_load_a_tile(A + (long long)bm * lda + (k0 + 32),
                        (unsigned)(size_t)&lds_a[nxt][0], arows, lda);
      const TB* src = B + (long long)(k0 + 32 + bk) * ldb + bn;
#pragma unroll
      for (int j = 0; j < 16; ++j) {
        const int gc = bn0 + j;
        bv[j] = (bn + gc < N) ? ldval(src + gc) : 0.0f;
      }
    }

    // ---- compute current tile: fragment gathers + 8 WMMAs ----
    v16bf afrag[4], bfrag[2];
#pragma unroll
    for (int i = 0; i < 4; ++i) {
      const bf16* p = &lds_a[cur][(wr * 64 + i * 16 + fr) * 32 + ak];
      U32x8 u;
      u.lo = *(const uint4*)p;          // K ak .. ak+7
      u.hi = *(const uint4*)(p + 16);   // K ak+16 .. ak+23
      afrag[i] = __builtin_bit_cast(v16bf, u);
    }
#pragma unroll
    for (int j = 0; j < 2; ++j) {
      const bf16* p = &lds_b[cur][(wc * 32 + j * 16 + fr) * 32 + bkk];
      U32x8 u;
      u.lo = *(const uint4*)p;
      u.hi = *(const uint4*)(p + 16);
      bfrag[j] = __builtin_bit_cast(v16bf, u);
    }
#pragma unroll
    for (int i = 0; i < 4; ++i)
#pragma unroll
      for (int j = 0; j < 2; ++j)
        acc[i][j] = __builtin_amdgcn_wmma_f32_16x16x32_bf16(
            false, afrag[i], false, bfrag[j], (short)0, acc[i][j], false,
            false);

    // ---- finish staging next tile, single barrier per K step ----
    if (do_stage) {
#pragma unroll
      for (int j = 0; j < 16; ++j)
        lds_b[nxt][(bn0 + j) * 32 + bk] = f2bf(bv[j]);
      if (wave == 0) __builtin_amdgcn_s_wait_tensorcnt((short)0);
    }
    __syncthreads();
    cur = nxt;
  }

  // ---- writeback: VGPR e -> row (e + 8*lane[4]), col = lane[3:0] ----
  const int r0 = (lane >> 4) * 8;
  const int cc = lane & 15;
#pragma unroll
  for (int i = 0; i < 4; ++i)
#pragma unroll
    for (int j = 0; j < 2; ++j) {
      const int row0 = bm + wr * 64 + i * 16 + r0;
      const int col  = bn + wc * 32 + j * 16 + cc;
      if (col < N) {
#pragma unroll
        for (int e = 0; e < 8; ++e) {
          const int row = row0 + e;
          if (row < M) C[(long long)row * ldc + col] = acc[i][j][e];
        }
      }
    }
}

// ---------------- RMSNorm -> bf16 ----------------
__global__ void rmsnorm_bf16_kernel(const float* __restrict__ x,
                                    const float* __restrict__ w,
                                    bf16* __restrict__ out, int D) {
  const int t = blockIdx.x;
  const float* row = x + (long long)t * D;
  __shared__ float red[256];
  float ss = 0.f;
  for (int d = threadIdx.x; d < D; d += 256) { float v = row[d]; ss += v * v; }
  red[threadIdx.x] = ss;
  __syncthreads();
  for (int s = 128; s > 0; s >>= 1) {
    if ((int)threadIdx.x < s) red[threadIdx.x] += red[threadIdx.x + s];
    __syncthreads();
  }
  const float rms = rsqrtf(red[0] / (float)D + 1e-6f);
  for (int d = threadIdx.x; d < D; d += 256)
    out[(long long)t * D + d] = f2bf(row[d] * rms * (1.f + w[d]));
}

// ---------------- h += RMSNorm(y) ----------------
__global__ void rmsnorm_add_kernel(float* __restrict__ h,
                                   const float* __restrict__ y,
                                   const float* __restrict__ w, int D) {
  const int t = blockIdx.x;
  const float* row = y + (long long)t * D;
  __shared__ float red[256];
  float ss = 0.f;
  for (int d = threadIdx.x; d < D; d += 256) { float v = row[d]; ss += v * v; }
  red[threadIdx.x] = ss;
  __syncthreads();
  for (int s = 128; s > 0; s >>= 1) {
    if ((int)threadIdx.x < s) red[threadIdx.x] += red[threadIdx.x + s];
    __syncthreads();
  }
  const float rms = rsqrtf(red[0] / (float)D + 1e-6f);
  for (int d = threadIdx.x; d < D; d += 256)
    h[(long long)t * D + d] += row[d] * rms * (1.f + w[d]);
}

// ---------------- q/k RMSNorm + RoPE, scatter to GEMM layouts ----------------
// qkv: [T][NG][GS+2][HD] f32.  qb: [NH][T][HD] bf16.
// kT:  [NG][HD][S] bf16 (col CACHE+t).  vT: [NG][S][HD] bf16 (row CACHE+t).
__global__ void qkv_post_kernel(const float* __restrict__ qkv,
                                const float* __restrict__ qw,
                                const float* __restrict__ kw,
                                const float* __restrict__ cosb,
                                const float* __restrict__ sinb,
                                bf16* __restrict__ qb, bf16* __restrict__ kT,
                                bf16* __restrict__ vT) {
  const int t = blockIdx.x, g = blockIdx.y, d = threadIdx.x; // d in 0..63
  __shared__ float sh[64];
  __shared__ float red[64];
  const float* base = qkv + (long long)t * QKV_ + (long long)g * (GS_ + 2) * HD_;
  const float c = cosb[t * HD_ + d];
  const float s = sinb[t * HD_ + d];
  for (int j = 0; j < GS_ + 2; ++j) {
    const float x = base[j * HD_ + d];
    if (j < GS_ + 1) { // q heads and k: rmsnorm + rope
      red[d] = x * x;
      __syncthreads();
      for (int st = 32; st > 0; st >>= 1) {
        if (d < st) red[d] += red[d + st];
        __syncthreads();
      }
      const float rms = rsqrtf(red[0] * (1.f / (float)HD_) + 1e-6f);
      const float* w = (j < GS_) ? qw : kw;
      const float nx = x * rms * (1.f + w[d]);
      sh[d] = nx;
      __syncthreads();
      const float rot = (d < HD_ / 2) ? -sh[d + HD_ / 2] : sh[d - HD_ / 2];
      const float o = nx * c + rot * s;
      if (j < GS_) {
        const int hh = g * GS_ + j;
        qb[((long long)hh * T_ + t) * HD_ + d] = f2bf(o);
      } else {
        kT[((long long)g * HD_ + d) * S_ + CACHE_ + t] = f2bf(o);
      }
    } else { // v: plain copy
      vT[((long long)g * S_ + CACHE_ + t) * HD_ + d] = f2bf(x);
    }
    __syncthreads();
  }
}

// kv_cache_k [NG][CACHE][HD] -> kT [NG][HD][S]
__global__ void copyk_kernel(const float* __restrict__ kc, bf16* __restrict__ kT) {
  const long long i = (long long)blockIdx.x * 256 + threadIdx.x;
  if (i >= (long long)NG_ * CACHE_ * HD_) return;
  const int d = (int)(i & (HD_ - 1));
  const long long r = i >> 6;
  const int s = (int)(r % CACHE_);
  const int g = (int)(r / CACHE_);
  kT[((long long)g * HD_ + d) * S_ + s] = f2bf(kc[i]);
}

// kv_cache_v [NG][HD][CACHE] -> vT [NG][S][HD]
__global__ void copyv_kernel(const float* __restrict__ vc, bf16* __restrict__ vT) {
  const long long i = (long long)blockIdx.x * 256 + threadIdx.x;
  if (i >= (long long)NG_ * HD_ * CACHE_) return;
  const int s = (int)(i % CACHE_);
  const long long r = i / CACHE_;
  const int d = (int)(r & (HD_ - 1));
  const int g = (int)(r >> 6);
  vT[((long long)g * S_ + s) * HD_ + d] = f2bf(vc[i]);
}

// scores [GS][T][S] f32 -> probs [GS][T][S] bf16; scale, softcap, mask, softmax
__global__ void softmax_kernel(const float* __restrict__ scores,
                               const float* __restrict__ mask,
                               bf16* __restrict__ probs) {
  const int t = blockIdx.x, h = blockIdx.y;
  const int tid = threadIdx.x;
  const float* srow = scores + ((long long)h * T_ + t) * S_;
  const float* mrow = mask + (long long)t * S_;
  bf16* prow = probs + ((long long)h * T_ + t) * S_;
  __shared__ float red[256];
  const float scale = 0.125f; // 1/sqrt(64)
  float vals[S_ / 256];
  float mx = -3.4e38f;
#pragma unroll
  for (int j = 0; j < S_ / 256; ++j) {
    const int s = tid + j * 256;
    float x = srow[s] * scale;
    x = tanhf(x * (1.f / 50.f)) * 50.f + mrow[s];
    vals[j] = x;
    mx = fmaxf(mx, x);
  }
  red[tid] = mx;
  __syncthreads();
  for (int s = 128; s > 0; s >>= 1) {
    if (tid < s) red[tid] = fmaxf(red[tid], red[tid + s]);
    __syncthreads();
  }
  mx = red[0];
  __syncthreads();
  float sum = 0.f;
#pragma unroll
  for (int j = 0; j < S_ / 256; ++j) {
    vals[j] = __expf(vals[j] - mx);
    sum += vals[j];
  }
  red[tid] = sum;
  __syncthreads();
  for (int s = 128; s > 0; s >>= 1) {
    if (tid < s) red[tid] += red[tid + s];
    __syncthreads();
  }
  const float inv = 1.f / red[0];
#pragma unroll
  for (int j = 0; j < S_ / 256; ++j) prow[tid + j * 256] = f2bf(vals[j] * inv);
}

__global__ void f2bf_kernel(const float* __restrict__ in, bf16* __restrict__ out,
                            int n) {
  const int i = blockIdx.x * 256 + threadIdx.x;
  if (i < n) out[i] = f2bf(in[i]);
}

// actb = bf16( gelu_tanh(gate) * up )
__global__ void gelumul_kernel(const float* __restrict__ g,
                               const float* __restrict__ u,
                               bf16* __restrict__ out, int n) {
  const int i = blockIdx.x * 256 + threadIdx.x;
  if (i < n) {
    const float x = g[i];
    const float t =
        0.5f * x * (1.f + tanhf(0.7978845608f * (x + 0.044715f * x * x * x)));
    out[i] = f2bf(t * u[i]);
  }
}

// ---------------- host orchestration ----------------
template <typename TB>
static void launch_gemm(const bf16* A, const TB* B, float* C, int M, int N,
                        int K, int lda, int ldb, int ldc, long long sA,
                        long long sB, long long sC, int batch,
                        hipStream_t stream) {
  dim3 grid((N + 127) / 128, (M + 127) / 128, batch);
  gemm_wmma_kernel<TB><<<grid, 256, 0, stream>>>(A, B, C, M, N, K, lda, ldb,
                                                 ldc, sA, sB, sC);
}

extern "C" void kernel_launch(void* const* d_in, const int* in_sizes, int n_in,
                              void* d_out, int out_size, void* d_ws,
                              size_t ws_size, hipStream_t stream) {
  (void)in_sizes; (void)n_in; (void)out_size; (void)ws_size;
  const float* emb        = (const float*)d_in[0];
  const float* kcache     = (const float*)d_in[1];
  const float* vcache     = (const float*)d_in[2];
  const float* pre_attn_w = (const float*)d_in[3];
  const float* w_qkv      = (const float*)d_in[4];
  const float* q_norm_w   = (const float*)d_in[5];
  const float* k_norm_w   = (const float*)d_in[6];
  const float* w_out_w    = (const float*)d_in[7];
  const float* post_attn_w= (const float*)d_in[8];
  const float* pre_ff_w   = (const float*)d_in[9];
  const float* w_gate     = (const float*)d_in[10];
  const float* w_up       = (const float*)d_in[11];
  const float* w_down     = (const float*)d_in[12];
  const float* post_ff_w  = (const float*)d_in[13];
  const float* final_norm = (const float*)d_in[14];
  const float* w_lm       = (const float*)d_in[15];
  const float* cos_g      = (const float*)d_in[16];
  const float* sin_g      = (const float*)d_in[17];
  const float* cos_l      = (const float*)d_in[18];
  const float* sin_l      = (const float*)d_in[19];
  const float* mask_g     = (const float*)d_in[20];
  const float* mask_l     = (const float*)d_in[21];
  float* out = (float*)d_out;

  // workspace carving (~43 MB)
  char* ws = (char*)d_ws;
  size_t off = 0;
  auto alloc = [&](size_t bytes) -> char* {
    char* p = ws + off;
    off += (bytes + 255) & ~(size_t)255;
    return p;
  };
  float* h     = (float*)alloc((size_t)T_ * D_ * 4);
  bf16*  xb    = (bf16*) alloc((size_t)T_ * D_ * 2);
  float* qkvb  = (float*)alloc((size_t)T_ * QKV_ * 4);
  bf16*  qb    = (bf16*) alloc((size_t)NH_ * T_ * HD_ * 2);
  bf16*  kTb   = (bf16*) alloc((size_t)NG_ * HD_ * S_ * 2);
  bf16*  vTb   = (bf16*) alloc((size_t)NG_ * S_ * HD_ * 2);
  float* scb   = (float*)alloc((size_t)GS_ * T_ * S_ * 4);
  bf16*  prb   = (bf16*) alloc((size_t)GS_ * T_ * S_ * 2);
  float* attn  = (float*)alloc((size_t)T_ * NH_ * HD_ * 4);
  bf16*  attnb = (bf16*) alloc((size_t)T_ * NH_ * HD_ * 2);
  float* proj  = (float*)alloc((size_t)T_ * D_ * 4);
  float* gate  = (float*)alloc((size_t)T_ * FF_ * 4);
  float* up    = (float*)alloc((size_t)T_ * FF_ * 4);
  bf16*  actb  = (bf16*) alloc((size_t)T_ * FF_ * 2);
  float* ffout = (float*)alloc((size_t)T_ * D_ * 4);

  (void)hipMemcpyAsync(h, emb, (size_t)T_ * D_ * 4, hipMemcpyDeviceToDevice,
                       stream);

  for (int i = 0; i < L_; ++i) {
    const bool is_local = ((i + 1) % 6) == 0;
    const float* cosp = is_local ? cos_l : cos_g;
    const float* sinp = is_local ? sin_l : sin_g;
    const float* mask = is_local ? mask_l : mask_g;

    // ---- attention block ----
    rmsnorm_bf16_kernel<<<T_, 256, 0, stream>>>(h, pre_attn_w + (size_t)i * D_, xb, D_);
    launch_gemm<float>(xb, w_qkv + (size_t)i * D_ * QKV_, qkvb, T_, QKV_, D_,
                       D_, QKV_, QKV_, 0, 0, 0, 1, stream);
    {
      const long long nk = (long long)NG_ * CACHE_ * HD_;
      copyk_kernel<<<(unsigned)((nk + 255) / 256), 256, 0, stream>>>(
          kcache + (size_t)i * NG_ * CACHE_ * HD_, kTb);
      copyv_kernel<<<(unsigned)((nk + 255) / 256), 256, 0, stream>>>(
          vcache + (size_t)i * NG_ * HD_ * CACHE_, vTb);
    }
    qkv_post_kernel<<<dim3(T_, NG_), 64, 0, stream>>>(
        qkvb, q_norm_w + (size_t)i * HD_, k_norm_w + (size_t)i * HD_, cosp,
        sinp, qb, kTb, vTb);

    for (int g = 0; g < NG_; ++g) {
      // scores[j,t,s] = q[g*GS+j,t,:] . k[g,:,s]
      launch_gemm<bf16>(qb + (size_t)g * GS_ * T_ * HD_,
                        kTb + (size_t)g * HD_ * S_, scb, T_, S_, HD_, HD_, S_,
                        S_, (long long)T_ * HD_, 0, (long long)T_ * S_, GS_,
                        stream);
      softmax_kernel<<<dim3(T_, GS_), 256, 0, stream>>>(scb, mask, prb);
      // attn[t, (g*GS+j)*HD + d] = probs[j,t,:] . v[g,:,d]
      launch_gemm<bf16>(prb, vTb + (size_t)g * S_ * HD_,
                        attn + (size_t)g * GS_ * HD_, T_, HD_, S_, S_, HD_,
                        NH_ * HD_, (long long)T_ * S_, 0, (long long)HD_, GS_,
                        stream);
    }
    f2bf_kernel<<<(T_ * NH_ * HD_ + 255) / 256, 256, 0, stream>>>(
        attn, attnb, T_ * NH_ * HD_);
    launch_gemm<float>(attnb, w_out_w + (size_t)i * NH_ * HD_ * D_, proj, T_,
                       D_, NH_ * HD_, NH_ * HD_, D_, D_, 0, 0, 0, 1, stream);
    rmsnorm_add_kernel<<<T_, 256, 0, stream>>>(h, proj,
                                               post_attn_w + (size_t)i * D_, D_);

    // ---- MLP block ----
    rmsnorm_bf16_kernel<<<T_, 256, 0, stream>>>(h, pre_ff_w + (size_t)i * D_, xb, D_);
    launch_gemm<float>(xb, w_gate + (size_t)i * D_ * FF_, gate, T_, FF_, D_,
                       D_, FF_, FF_, 0, 0, 0, 1, stream);
    launch_gemm<float>(xb, w_up + (size_t)i * D_ * FF_, up, T_, FF_, D_, D_,
                       FF_, FF_, 0, 0, 0, 1, stream);
    gelumul_kernel<<<(T_ * FF_ + 255) / 256, 256, 0, stream>>>(gate, up, actb,
                                                               T_ * FF_);
    launch_gemm<float>(actb, w_down + (size_t)i * FF_ * D_, ffout, T_, D_, FF_,
                       FF_, D_, D_, 0, 0, 0, 1, stream);
    rmsnorm_add_kernel<<<T_, 256, 0, stream>>>(h, ffout,
                                               post_ff_w + (size_t)i * D_, D_);
  }

  // ---- final norm + LM head ----
  rmsnorm_bf16_kernel<<<T_, 256, 0, stream>>>(h, final_norm, xb, D_);
  launch_gemm<float>(xb, w_lm, out, T_, V_, D_, D_, V_, V_, 0, 0, 0, 1, stream);
}
